// MultiClassDiceLoss_86732569575997
// MI455X (gfx1250) — compile-verified, matrix operand order checked by reading, not run on hardware
//
#include <hip/hip_runtime.h>

// ext-vector types (HIP's uint4 is a struct; __builtin_nontemporal_load needs a real vector)
typedef _Float16     v16h  __attribute__((ext_vector_type(16)));
typedef float        v8f   __attribute__((ext_vector_type(8)));
typedef unsigned int u32x4 __attribute__((ext_vector_type(4)));

// Bit patterns of 1.0f, 2.0f, 3.0f — the label values are exact small-int floats.
__device__ __forceinline__ void proc_pair(unsigned u1, unsigned u2, int* c) {
    const unsigned LB[3] = {0x3f800000u, 0x40000000u, 0x40400000u};
    const bool e12 = (u1 == u2);
#pragma unroll
    for (int l = 0; l < 3; ++l) {
        const bool a = (u1 == LB[l]);
        const bool b = (u2 == LB[l]);
        c[l]     += (int)a;           // |A_l| partial
        c[3 + l] += (int)b;           // |B_l| partial
        c[6 + l] += (int)(a && e12);  // |A_l ∩ B_l| partial
    }
}

__device__ __forceinline__ void proc_vec(u32x4 a, u32x4 b, int* c) {
    proc_pair(a.x, b.x, c);
    proc_pair(a.y, b.y, c);
    proc_pair(a.z, b.z, c);
    proc_pair(a.w, b.w, c);
}

__global__ void dice_init(int* __restrict__ gcnt) {
    if (threadIdx.x < 9) gcnt[threadIdx.x] = 0;
}

__global__ __launch_bounds__(256)
void dice_count(const u32x4* __restrict__ A, const u32x4* __restrict__ B,
                const unsigned* __restrict__ tailA, const unsigned* __restrict__ tailB,
                int ntail, int* __restrict__ gcnt, long long nvec) {
    int cnt[9] = {0, 0, 0, 0, 0, 0, 0, 0, 0};

    const long long tid    = (long long)blockIdx.x * blockDim.x + threadIdx.x;
    const long long stride = (long long)gridDim.x * blockDim.x;

    // Streaming phase: NT 128-bit loads, register accumulation.
    // 4x unrolled so 8 independent global_load_b128 are in flight per wave
    // before any consumer -> better HBM pipelining per wave.
    long long i = tid;
    for (; i + 3 * stride < nvec; i += 4 * stride) {
        u32x4 a0 = __builtin_nontemporal_load(&A[i]);
        u32x4 b0 = __builtin_nontemporal_load(&B[i]);
        u32x4 a1 = __builtin_nontemporal_load(&A[i + stride]);
        u32x4 b1 = __builtin_nontemporal_load(&B[i + stride]);
        u32x4 a2 = __builtin_nontemporal_load(&A[i + 2 * stride]);
        u32x4 b2 = __builtin_nontemporal_load(&B[i + 2 * stride]);
        u32x4 a3 = __builtin_nontemporal_load(&A[i + 3 * stride]);
        u32x4 b3 = __builtin_nontemporal_load(&B[i + 3 * stride]);
        proc_vec(a0, b0, cnt);
        proc_vec(a1, b1, cnt);
        proc_vec(a2, b2, cnt);
        proc_vec(a3, b3, cnt);
    }
    for (; i < nvec; i += stride) {
        u32x4 a = __builtin_nontemporal_load(&A[i]);
        u32x4 b = __builtin_nontemporal_load(&B[i]);
        proc_vec(a, b, cnt);
    }
    if (tid == 0) {
        for (int t = 0; t < ntail; ++t) proc_pair(tailA[t], tailB[t], cnt);
    }

    // Block reduction. Per-thread counters <= 259 < 2048 -> exact in f16.
    __shared__ _Float16 lds[256 * 16];
    const int t = threadIdx.x;
#pragma unroll
    for (int k = 0; k < 9; ++k) lds[t * 16 + k] = (_Float16)cnt[k];
#pragma unroll
    for (int k = 9; k < 16; ++k) lds[t * 16 + k] = (_Float16)0.0f;
    __syncthreads();

    // Wave 0 (EXEC all-ones across lanes 0..31) reduces 256x16 via 8 chained WMMAs:
    // D = ones(16x32) * B(32x16) + C  ->  every D row = column sums, exact in f32.
    if (t < 32) {
        const int col = t & 15;        // N = lane % 16
        const int g   = (t >> 4) & 1;  // lane group selects 16 of the 32 rows per chunk
        v16h ones;
#pragma unroll
        for (int s = 0; s < 16; ++s) ones[s] = (_Float16)1.0f;
        v8f acc = {};
#pragma unroll
        for (int chunk = 0; chunk < 8; ++chunk) {  // 8 chunks x 32 threads = 256
            v16h bm;
#pragma unroll
            for (int s = 0; s < 16; ++s)
                bm[s] = lds[(chunk * 32 + g * 16 + s) * 16 + col];
            // A = ones -> the per-lane K-slot ordering is irrelevant; only N=lane%16 matters.
            acc = __builtin_amdgcn_wmma_f32_16x16x32_f16(
                /*neg_a=*/false, ones, /*neg_b=*/false, bm,
                /*c_mod=*/(short)0, acc, /*reuse_a=*/false, /*reuse_b=*/false);
        }
        if (t < 9) atomicAdd(&gcnt[t], (int)(acc[0] + 0.5f));  // exact int in f32
    }
}

__global__ void dice_final(const int* __restrict__ gcnt, float* __restrict__ out) {
    if (threadIdx.x == 0 && blockIdx.x == 0) {
        const float eps = 1.1920928955078125e-7f;  // FLT_EPSILON, matches np.finfo(f32).eps
        float s = 0.0f;
#pragma unroll
        for (int l = 0; l < 3; ++l) {
            const float top = 2.0f * (float)gcnt[6 + l];
            const float bot = (float)gcnt[l] + (float)gcnt[3 + l];
            s += top / (bot + eps);
        }
        out[0] = 1.0f - s / 3.0f;
    }
}

extern "C" void kernel_launch(void* const* d_in, const int* in_sizes, int n_in,
                              void* d_out, int out_size, void* d_ws, size_t ws_size,
                              hipStream_t stream) {
    const unsigned* v1 = (const unsigned*)d_in[0];  // float32 label map, compared bitwise
    const unsigned* v2 = (const unsigned*)d_in[1];
    const long long n    = (long long)in_sizes[0];  // 512^3
    const long long nvec = n >> 2;
    const int       ntail = (int)(n & 3);

    int*   gcnt = (int*)d_ws;       // 9 global counters in workspace
    float* out  = (float*)d_out;

    dice_init<<<1, 32, 0, stream>>>(gcnt);
    dice_count<<<2048, 256, 0, stream>>>((const u32x4*)v1, (const u32x4*)v2,
                                         v1 + nvec * 4, v2 + nvec * 4, ntail,
                                         gcnt, nvec);
    dice_final<<<1, 32, 0, stream>>>(gcnt, out);
}
